// STBlockImproved_64261300683055
// MI455X (gfx1250) — matrix operand order repaired and checked by reading, not compiled, not run on hardware
//
#include <hip/hip_runtime.h>
#include <math.h>

// ---------------------------------------------------------------------------
// STBlock forward for MI455X (gfx1250, wave32).
// Heavy op: za = A(883x883) @ z(883x768) per batch, 49 times -> bf16 WMMA
// (v_wmma_f32_16x16x32_bf16, f32 accum). Memory-bound regime (23.3 TB/s).
// GEMM: 64x64 register blocking (16 accumulators), double-buffered K-loop so
// VMEM overlaps the matrix pipe; K-major bf16 Zt copy makes every fragment a
// contiguous 16B/32B global load (K order k = 16*hi + s, same in A and B).
// Input flattening (setup_inputs insertion order, nested dicts sorted-key):
//  0:X 1:A_hat | tcn1: 2..10 bb[9], 11..19 bw[9], 20 db, 21 dw, 22 pb, 23 pw
//  ode: 24 alpha, 25 d, 26 d2, 27 w, 28 w2
//  adp: 29 E1, 30 E2, 31 g0, 32 g1, 33 theta
//  fuse: 34 b1, 35 b2, 36 w1, 37 w2
//  tcn2: 38..46 bb, 47..55 bw, 56 pb, 57 pw
//  se: 58 b1, 59 b2, 60 w1, 61 w2 | bn: 62 beta, 63 gamma
// ---------------------------------------------------------------------------

typedef __bf16 bf16;
typedef __attribute__((ext_vector_type(8)))  __bf16 bf16x8;
typedef __attribute__((ext_vector_type(16))) __bf16 bf16x16;
typedef __attribute__((ext_vector_type(8)))  float  f32x8;

#define BB   32
#define NN   883
#define TT   12
#define CH   64
#define FF   768              // TT*CH, per-(b,n) feature row
#define NP   896              // padded node dim (28*32) for WMMA K / M tiles
#define MG   14               // NP/64 M-tile groups (64 rows per wave)
#define FS   12               // FF/64 f-strips (64 cols per wave)
#define NT_  (NN*TT)          // 10596
#define ZSZ  21700608ull      // BB*NN*FF

__device__ __constant__ int c_kk[9] = {2,2,2,3,3,3,5,5,5};
__device__ __constant__ int c_dd[9] = {1,2,4,1,2,4,1,2,4};

struct MstcW {
  const float* bw[9]; const float* bb[9];
  const float* pw; const float* pb;
  const float* dw; const float* db;
};

static __device__ __forceinline__ float sigmoidf_(float x) { return 1.f/(1.f+__expf(-x)); }
static __device__ __forceinline__ float clip01(float x)    { return fminf(fmaxf(x,0.f),1.f); }

// ---------------- prep: W = (w*clip(d))@w^T, W2, sigmoid(alpha) -------------
__global__ __launch_bounds__(256)
void k_prep(const float* __restrict__ alpha, const float* __restrict__ dv,
            const float* __restrict__ d2v, const float* __restrict__ w,
            const float* __restrict__ w2, float* __restrict__ Wm,
            float* __restrict__ W2m, float* __restrict__ sigA) {
  int tid = threadIdx.x;
  for (int e = tid; e < CH*CH; e += 256) {
    int a = e >> 6, c = e & 63; float s = 0.f;
    for (int k = 0; k < CH; ++k) s += w[a*CH+k]*clip01(dv[k])*w[c*CH+k];
    Wm[e] = s;
  }
  for (int e = tid; e < TT*TT; e += 256) {
    int a = e/TT, c = e - a*TT; float s = 0.f;
    for (int k = 0; k < TT; ++k) s += w2[a*TT+k]*clip01(d2v[k])*w2[c*TT+k];
    W2m[e] = s;
  }
  for (int n = tid; n < NN; n += 256) sigA[n] = sigmoidf_(alpha[n]);
}

// ---------------- A_hat -> padded bf16 --------------------------------------
__global__ void k_convA(const float* __restrict__ Ah, bf16* __restrict__ Abf) {
  for (int e = blockIdx.x*blockDim.x + threadIdx.x; e < NP*NP;
       e += gridDim.x*blockDim.x) {
    int i = e / NP, j = e - i*NP;
    float v = (i < NN && j < NN) ? Ah[(size_t)i*NN + j] : 0.f;
    Abf[e] = (bf16)v;
  }
}

// ---------------- adaptive adjacency row (relu->softmax->mix) ---------------
__global__ __launch_bounds__(256)
void k_adp_row(const float* __restrict__ Ah, const float* __restrict__ E1,
               const float* __restrict__ E2, const float* __restrict__ g0p,
               const float* __restrict__ g1p, bf16* __restrict__ A2) {
  __shared__ float sE1[16]; __shared__ float sRow[NN]; __shared__ float sRed[256];
  int i = blockIdx.x, tid = threadIdx.x;
  if (i >= NN) { for (int j = tid; j < NP; j += 256) A2[(size_t)i*NP+j] = (bf16)0.f; return; }
  if (tid < 16) sE1[tid] = E1[i*16 + tid];
  __syncthreads();
  float lmax = -3.4e38f;
  for (int j = tid; j < NN; j += 256) {
    const float* e2 = E2 + j*16; float s = 0.f;
    #pragma unroll
    for (int k = 0; k < 16; ++k) s += sE1[k]*e2[k];
    s = fmaxf(s, 0.f); sRow[j] = s; lmax = fmaxf(lmax, s);
  }
  sRed[tid] = lmax; __syncthreads();
  for (int s = 128; s > 0; s >>= 1) { if (tid < s) sRed[tid] = fmaxf(sRed[tid], sRed[tid+s]); __syncthreads(); }
  float mx = sRed[0]; __syncthreads();
  float lsum = 0.f;
  for (int j = tid; j < NN; j += 256) { float ev = __expf(sRow[j]-mx); sRow[j] = ev; lsum += ev; }
  sRed[tid] = lsum; __syncthreads();
  for (int s = 128; s > 0; s >>= 1) { if (tid < s) sRed[tid] += sRed[tid+s]; __syncthreads(); }
  float inv = 1.f / sRed[0];
  float sg0 = sigmoidf_(g0p[0]), sg1 = sigmoidf_(g1p[0]);
  for (int j = tid; j < NP; j += 256) {
    float v = (j < NN) ? (sg0*Ah[(size_t)i*NN+j] + sg1*sRow[j]*inv) : 0.f;
    A2[(size_t)i*NP + j] = (bf16)v;
  }
}

// ---------------- MSTC block (branch convs + pointwise), LDS staged ---------
template<int CIN, bool DOWN>
__global__ __launch_bounds__(256)
void k_mstc(const float* __restrict__ in, float* __restrict__ out, MstcW w) {
  __shared__ float sIn[TT*CIN];
  __shared__ float sFeat[TT*576];
  int tid = threadIdx.x;
  size_t ibase = (size_t)blockIdx.x * TT * CIN;
  for (int i = tid; i < TT*CIN; i += 256) sIn[i] = in[ibase + i];
  __syncthreads();
  for (int e = tid; e < TT*576; e += 256) {
    int t = e / 576, f = e - t*576;
    int i = f >> 6, o = f & 63;
    int k = c_kk[i], dl = c_dd[i];
    const float* wb = w.bw[i] + (size_t)o*CIN*k;
    float a = w.bb[i][o];
    for (int tap = 0; tap < k; ++tap) {
      int ts = t - (k-1-tap)*dl;
      if (ts < 0) continue;
      const float* xr = sIn + ts*CIN;
      for (int ci = 0; ci < CIN; ++ci) a += wb[ci*k + tap]*xr[ci];
    }
    sFeat[e] = fmaxf(a, 0.f);
  }
  __syncthreads();
  size_t obase = (size_t)blockIdx.x * FF;
  for (int e = tid; e < FF; e += 256) {
    int t = e >> 6, o = e & 63;
    const float* fr = sFeat + t*576;
    const float* pr = w.pw + (size_t)o*576;
    float a = w.pb[o];
    #pragma unroll 8
    for (int f = 0; f < 576; ++f) a += pr[f]*fr[f];
    float r = fmaxf(a, 0.f);
    if (DOWN) {
      float da = w.db[o];
      for (int ci = 0; ci < CIN; ++ci) da += w.dw[o*CIN+ci]*sIn[t*CIN+ci];
      r += da;
    }
    out[obase + e] = fmaxf(r, 0.f);
  }
}

// ---------------- h -> z copy + bf16 K-major transpose Zt[b][f][j] ----------
__global__ void k_ztinit(const float* __restrict__ h, float* __restrict__ z,
                         bf16* __restrict__ Zt) {
  size_t stride = (size_t)gridDim.x * blockDim.x;
  for (size_t e = (size_t)blockIdx.x*blockDim.x + threadIdx.x; e < ZSZ; e += stride) {
    float v = h[e];
    z[e] = v;
    size_t row = e / FF; int f = (int)(e - row*FF);
    int b = (int)(row / NN); int n = (int)(row - (size_t)b*NN);
    Zt[((size_t)b*FF + f)*NP + n] = (bf16)v;
  }
}

// ---------------- WMMA GEMM fragment helpers --------------------------------
static __device__ __forceinline__
void load_a4(const bf16* __restrict__ aBase, int kt, bf16x16 (&a)[4]) {
  #pragma unroll
  for (int it = 0; it < 4; ++it) {
    const bf16* ak = aBase + (size_t)it*16*NP + (size_t)kt*32;
    bf16x8 lo = *(const bf16x8*)(ak);
    bf16x8 hi = *(const bf16x8*)(ak + 8);
    a[it] = __builtin_shufflevector(lo, hi,
        0,1,2,3,4,5,6,7,8,9,10,11,12,13,14,15);
  }
}
static __device__ __forceinline__
void load_b4(const bf16* __restrict__ zBase, int kt, bf16x16 (&bv)[4]) {
  #pragma unroll
  for (int ft = 0; ft < 4; ++ft)
    bv[ft] = *(const bf16x16*)(zBase + (size_t)ft*16*NP + (size_t)kt*32);
}
static __device__ __forceinline__
void mma16(const bf16x16 (&a)[4], const bf16x16 (&bv)[4], f32x8 (&acc)[4][4]) {
  #pragma unroll
  for (int it = 0; it < 4; ++it)
    #pragma unroll
    for (int ft = 0; ft < 4; ++ft)
      acc[it][ft] = __builtin_amdgcn_wmma_f32_16x16x32_bf16(
          false, a[it], false, bv[ft], (short)0, acc[it][ft], false, false);
}

// ---------------- WMMA GEMM: out[b][i][f] = sum_j A[i,j] * Zt[b][f][j] ------
// One wave per 64(M) x 64(F) block: 16 accumulators, A frags reused across 4
// F-tiles and B frags across 4 M-tiles (4x less fetch traffic than 16x64).
// K-loop double-buffered so global loads overlap the WMMA pipe.
__global__ __launch_bounds__(128)
void k_wmma_gemm(const bf16* __restrict__ Abf, const bf16* __restrict__ Zt,
                 float* __restrict__ out) {
  int lane = threadIdx.x & 31;
  int wid  = blockIdx.x * 4 + (threadIdx.x >> 5);
  int b    = wid / (MG*FS);
  int rem  = wid - b*(MG*FS);
  int mg   = rem / FS;        // 64-row M group
  int fs   = rem - mg*FS;     // 64-col F strip
  int m  = lane & 15;
  int hi = lane >> 4;
  // A fragment (16-bit 16x32): lane m, halves 0..15 = K in [hi*16, hi*16+16)
  // -> two adjacent 16B runs (matches B's K order).
  const bf16* aBase = Abf + (size_t)(mg*64 + m)*NP + hi*16;
  // B fragment (32x16): lane col = m, halves 0..15 = K in [hi*16, hi*16+16)
  // -> one contiguous 32B run along the K-major Zt row.
  const bf16* zBase = Zt + ((size_t)b*FF + fs*64 + m)*NP + hi*16;

  f32x8 acc[4][4] = {};
  bf16x16 a0[4], a1[4], b0[4], b1[4];
  load_a4(aBase, 0, a0);
  load_b4(zBase, 0, b0);
  for (int kt = 0; kt < 28; kt += 2) {
    load_a4(aBase, kt+1, a1);
    load_b4(zBase, kt+1, b1);
    mma16(a0, b0, acc);
    if (kt + 2 < 28) {
      load_a4(aBase, kt+2, a0);
      load_b4(zBase, kt+2, b0);
    }
    mma16(a1, b1, acc);
  }
  // D layout: VGPR r -> M = r + hi*8, N = lane&15
  #pragma unroll
  for (int it = 0; it < 4; ++it) {
    int row0 = mg*64 + it*16 + hi*8;
    #pragma unroll
    for (int ft = 0; ft < 4; ++ft) {
      int col = fs*64 + m + ft*16;
      #pragma unroll
      for (int r = 0; r < 8; ++r) {
        int row = row0 + r;
        if (row < NN) out[((size_t)b*NN + row)*FF + col] = acc[it][ft][r];
      }
    }
  }
}

// ---------------- theta: y <- relu(y @ theta), in place (LDS staged) --------
__global__ __launch_bounds__(256)
void k_theta(float* __restrict__ y, const float* __restrict__ theta) {
  __shared__ float sY[FF];
  int tid = threadIdx.x;
  size_t base = (size_t)blockIdx.x * FF;
  for (int i = tid; i < FF; i += 256) sY[i] = y[base + i];
  __syncthreads();
  for (int e = tid; e < FF; e += 256) {
    int t = e >> 6, c = e & 63;
    const float* zr = sY + t*CH;
    float s = 0.f;
    #pragma unroll 8
    for (int d = 0; d < CH; ++d) s += zr[d]*theta[d*CH + c];
    y[base + e] = fmaxf(s, 0.f);
  }
}

// ---------------- fused RK4 stage: f(z)=0.5*sig(a)*za -3z +zW +W2'z +x0 -----
__global__ __launch_bounds__(256)
void k_ode_stage(const float* __restrict__ z, const float* __restrict__ zstage,
                 const float* __restrict__ za, const float* __restrict__ x0,
                 float* __restrict__ acc, float* __restrict__ zsout,
                 bf16* __restrict__ Zt, const float* __restrict__ sigA,
                 const float* __restrict__ Wm, const float* __restrict__ W2m,
                 float dt, int stage) {
  __shared__ float sZ[FF];
  __shared__ float sW[CH*CH];
  __shared__ float sW2[TT*TT];
  int tid = threadIdx.x;
  int row = blockIdx.x;                 // b*NN + n
  int b = row / NN, n = row - b*NN;
  size_t base = (size_t)row * FF;
  for (int i = tid; i < FF; i += 256)     sZ[i] = zstage[base + i];
  for (int i = tid; i < CH*CH; i += 256)  sW[i] = Wm[i];
  if (tid < TT*TT) sW2[tid] = W2m[tid];
  __syncthreads();
  float sa = sigA[n] * 0.5f;
  for (int e = tid; e < FF; e += 256) {
    int t = e >> 6, c = e & 63;
    const float* zr = sZ + t*CH;
    const float* wc = sW + c;
    float zw = 0.f;
    #pragma unroll 8
    for (int d = 0; d < CH; ++d) zw += zr[d]*wc[d*CH];
    float zw2 = 0.f;
    #pragma unroll
    for (int tt = 0; tt < TT; ++tt) zw2 += sW2[tt*TT + t]*sZ[tt*CH + c];
    size_t off = base + e;
    float fv = sa*za[off] - 3.0f*sZ[e] + zw + zw2 + x0[off];
    float zs;
    if      (stage == 0) { acc[off] = fv;          zs = z[off] + 0.5f*dt*fv; }
    else if (stage == 1) { acc[off] += 2.f*fv;     zs = z[off] + 0.5f*dt*fv; }
    else if (stage == 2) { acc[off] += 2.f*fv;     zs = z[off] + dt*fv; }
    else                 { zs = z[off] + (dt*(1.f/6.f))*(acc[off] + fv); }
    zsout[off] = zs;
    Zt[((size_t)b*FF + e)*NP + n] = (bf16)zs;
  }
}

// ---------------- mean over (N,T) per (b,c) ---------------------------------
__global__ __launch_bounds__(256)
void k_meanNT(const float* __restrict__ src, float* __restrict__ dst,
              int stride, int off, int doRelu) {
  __shared__ float sR[256];
  int b = blockIdx.x >> 6, c = blockIdx.x & 63;
  float s = 0.f;
  for (int nt = threadIdx.x; nt < NT_; nt += 256) {
    float v = src[((size_t)b*NT_ + nt)*CH + c];
    if (doRelu) v = fmaxf(v, 0.f);
    s += v;
  }
  sR[threadIdx.x] = s; __syncthreads();
  for (int st = 128; st > 0; st >>= 1) { if (threadIdx.x < st) sR[threadIdx.x] += sR[threadIdx.x+st]; __syncthreads(); }
  if (threadIdx.x == 0) dst[b*stride + off + c] = sR[0]*(1.f/(float)NT_);
}

// ---------------- fuse gating MLP (per batch) -------------------------------
__global__ void k_fuse_mlp(const float* __restrict__ pooled, const float* __restrict__ w1,
                           const float* __restrict__ b1, const float* __restrict__ w2,
                           const float* __restrict__ b2, float* __restrict__ gbuf) {
  __shared__ float sh[64];
  int b = blockIdx.x, o = threadIdx.x;
  float s = b1[o];
  for (int i = 0; i < 128; ++i) s += w1[o*128 + i]*pooled[b*128 + i];
  sh[o] = fmaxf(s, 0.f);
  __syncthreads();
  float g = b2[o];
  for (int d = 0; d < 64; ++d) g += w2[o*64 + d]*sh[d];
  gbuf[b*64 + o] = sigmoidf_(g);
}

// ---------------- SE MLP (per batch) ----------------------------------------
__global__ void k_se_mlp(const float* __restrict__ pooled, const float* __restrict__ w1,
                         const float* __restrict__ b1, const float* __restrict__ w2,
                         const float* __restrict__ b2, float* __restrict__ g2) {
  __shared__ float sh[8];
  int b = blockIdx.x, tid = threadIdx.x;
  if (tid < 8) {
    float s = b1[tid];
    for (int c = 0; c < 64; ++c) s += w1[tid*64 + c]*pooled[b*64 + c];
    sh[tid] = fmaxf(s, 0.f);
  }
  __syncthreads();
  float g = b2[tid];
  for (int j = 0; j < 8; ++j) g += w2[tid*8 + j]*sh[j];
  g2[b*64 + tid] = sigmoidf_(g);
}

// ---------------- gate: relu(g*relu(y1) + (1-g)*y2) -------------------------
__global__ void k_gate(const float* __restrict__ y1, const float* __restrict__ y2,
                       const float* __restrict__ gbuf, float* __restrict__ out) {
  size_t stride = (size_t)gridDim.x * blockDim.x;
  for (size_t e = (size_t)blockIdx.x*blockDim.x + threadIdx.x; e < ZSZ; e += stride) {
    int c = (int)(e & 63);
    int b = (int)(e / ((size_t)NT_*CH));
    float g = gbuf[b*64 + c];
    float v = g*fmaxf(y1[e], 0.f) + (1.f - g)*y2[e];
    out[e] = fmaxf(v, 0.f);
  }
}

// ---------------- BN stats over (B,T,C) per node n --------------------------
__global__ __launch_bounds__(256)
void k_bn_stats(const float* __restrict__ y, const float* __restrict__ g2,
                float* __restrict__ stats) {
  __shared__ float sS[256], sQ[256];
  int n = blockIdx.x, tid = threadIdx.x;
  float s = 0.f, q = 0.f;
  for (int idx = tid; idx < BB*FF; idx += 256) {
    int b = idx / FF, r = idx - b*FF, c = r & 63;
    float v = y[((size_t)b*NN + n)*FF + r] * g2[b*64 + c];
    s += v; q += v*v;
  }
  sS[tid] = s; sQ[tid] = q; __syncthreads();
  for (int st = 128; st > 0; st >>= 1) {
    if (tid < st) { sS[tid] += sS[tid+st]; sQ[tid] += sQ[tid+st]; }
    __syncthreads();
  }
  if (tid == 0) {
    float m = sS[0]*(1.f/(BB*FF));
    stats[n*2] = m;
    stats[n*2+1] = sQ[0]*(1.f/(BB*FF)) - m*m;
  }
}

__global__ void k_bn_apply(const float* __restrict__ y, const float* __restrict__ g2,
                           const float* __restrict__ stats, const float* __restrict__ gamma,
                           const float* __restrict__ beta, float* __restrict__ out) {
  size_t stride = (size_t)gridDim.x * blockDim.x;
  for (size_t e = (size_t)blockIdx.x*blockDim.x + threadIdx.x; e < ZSZ; e += stride) {
    size_t row = e / FF;
    int n = (int)(row % NN);
    int b = (int)(row / NN);
    int c = (int)(e & 63);
    float v = y[e]*g2[b*64 + c];
    float m = stats[n*2], var = stats[n*2+1];
    out[e] = gamma[n]*(v - m)*rsqrtf(var + 1e-5f) + beta[n];
  }
}

// ---------------------------------------------------------------------------
extern "C" void kernel_launch(void* const* d_in, const int* in_sizes, int n_in,
                              void* d_out, int out_size, void* d_ws, size_t ws_size,
                              hipStream_t stream) {
  (void)in_sizes; (void)n_in; (void)out_size; (void)ws_size;
  const float* X    = (const float*)d_in[0];
  const float* Ahat = (const float*)d_in[1];
  MstcW tw1, tw2;
  for (int i = 0; i < 9; ++i) { tw1.bb[i] = (const float*)d_in[2+i];  tw1.bw[i] = (const float*)d_in[11+i]; }
  tw1.db = (const float*)d_in[20]; tw1.dw = (const float*)d_in[21];
  tw1.pb = (const float*)d_in[22]; tw1.pw = (const float*)d_in[23];
  const float* alpha = (const float*)d_in[24];
  const float* dv    = (const float*)d_in[25];
  const float* d2v   = (const float*)d_in[26];
  const float* wI    = (const float*)d_in[27];
  const float* w2I   = (const float*)d_in[28];
  const float* E1    = (const float*)d_in[29];
  const float* E2    = (const float*)d_in[30];
  const float* g0p   = (const float*)d_in[31];
  const float* g1p   = (const float*)d_in[32];
  const float* theta = (const float*)d_in[33];
  const float* fb1   = (const float*)d_in[34];
  const float* fb2   = (const float*)d_in[35];
  const float* fw1   = (const float*)d_in[36];
  const float* fw2   = (const float*)d_in[37];
  for (int i = 0; i < 9; ++i) { tw2.bb[i] = (const float*)d_in[38+i]; tw2.bw[i] = (const float*)d_in[47+i]; }
  tw2.pb = (const float*)d_in[56]; tw2.pw = (const float*)d_in[57];
  tw2.db = nullptr; tw2.dw = nullptr;
  const float* sb1   = (const float*)d_in[58];
  const float* sb2   = (const float*)d_in[59];
  const float* sw1   = (const float*)d_in[60];
  const float* sw2   = (const float*)d_in[61];
  const float* beta  = (const float*)d_in[62];
  const float* gamma = (const float*)d_in[63];

  // workspace carve-up (~568 MB)
  char* p = (char*)d_ws;
  float* h    = (float*)p; p += ZSZ*4;          // x0 (kept through ODE)
  float* z    = (float*)p; p += ZSZ*4;
  float* zs   = (float*)p; p += ZSZ*4;          // RK4 stage input / fused y
  float* acc  = (float*)p; p += ZSZ*4;          // RK4 accum / mstc2 out
  float* za   = (float*)p; p += ZSZ*4;
  float* y2b  = (float*)p; p += ZSZ*4;
  bf16* Zt    = (bf16*)p;  p += (size_t)BB*FF*NP*2;
  bf16* Abf   = (bf16*)p;  p += (size_t)NP*NP*2;
  bf16* A2bf  = (bf16*)p;  p += (size_t)NP*NP*2;
  float* Wm   = (float*)p; p += 4096*4;
  float* W2m  = (float*)p; p += 256*4;
  float* sigA = (float*)p; p += 896*4;
  float* pooled  = (float*)p; p += BB*128*4;
  float* gbuf    = (float*)p; p += BB*64*4;
  float* pooled2 = (float*)p; p += BB*64*4;
  float* g2      = (float*)p; p += BB*64*4;
  float* stats   = (float*)p; p += 2048*4;

  const int NBLK = BB*NN;                 // 28256 (b,n) blocks
  const int GEMM_BLK = BB*MG*FS/4;        // 1344 blocks of 4 waves (64x64 tiles)

  k_prep<<<1,256,0,stream>>>(alpha, dv, d2v, wI, w2I, Wm, W2m, sigA);
  k_convA<<<1024,256,0,stream>>>(Ahat, Abf);
  k_mstc<3,true><<<NBLK,256,0,stream>>>(X, h, tw1);
  k_ztinit<<<4096,256,0,stream>>>(h, z, Zt);
  // adaptive GCN branch (uses Zt(h) before ODE overwrites it)
  k_adp_row<<<NP,256,0,stream>>>(Ahat, E1, E2, g0p, g1p, A2bf);
  k_wmma_gemm<<<GEMM_BLK,128,0,stream>>>(A2bf, Zt, y2b);
  k_theta<<<NBLK,256,0,stream>>>(y2b, theta);
  // RK4 x 12
  const float dt = 1.0f/12.0f;
  for (int s = 0; s < 12; ++s) {
    k_wmma_gemm<<<GEMM_BLK,128,0,stream>>>(Abf, Zt, za);
    k_ode_stage<<<NBLK,256,0,stream>>>(z, z,  za, h, acc, zs, Zt, sigA, Wm, W2m, dt, 0);
    k_wmma_gemm<<<GEMM_BLK,128,0,stream>>>(Abf, Zt, za);
    k_ode_stage<<<NBLK,256,0,stream>>>(z, zs, za, h, acc, zs, Zt, sigA, Wm, W2m, dt, 1);
    k_wmma_gemm<<<GEMM_BLK,128,0,stream>>>(Abf, Zt, za);
    k_ode_stage<<<NBLK,256,0,stream>>>(z, zs, za, h, acc, zs, Zt, sigA, Wm, W2m, dt, 2);
    k_wmma_gemm<<<GEMM_BLK,128,0,stream>>>(Abf, Zt, za);
    k_ode_stage<<<NBLK,256,0,stream>>>(z, zs, za, h, acc, z,  Zt, sigA, Wm, W2m, dt, 3);
  }
  // fuse(y1=relu(z), y2)
  k_meanNT<<<BB*64,256,0,stream>>>(z,   pooled, 128, 0,  1);
  k_meanNT<<<BB*64,256,0,stream>>>(y2b, pooled, 128, 64, 0);
  k_fuse_mlp<<<BB,64,0,stream>>>(pooled, fw1, fb1, fw2, fb2, gbuf);
  k_gate<<<4096,256,0,stream>>>(z, y2b, gbuf, zs);
  // second MSTC, SE, BN
  k_mstc<64,false><<<NBLK,256,0,stream>>>(zs, acc, tw2);
  k_meanNT<<<BB*64,256,0,stream>>>(acc, pooled2, 64, 0, 0);
  k_se_mlp<<<BB,64,0,stream>>>(pooled2, sw1, sb1, sw2, sb2, g2);
  k_bn_stats<<<NN,256,0,stream>>>(acc, g2, stats);
  k_bn_apply<<<4096,256,0,stream>>>(acc, g2, stats, gamma, beta, (float*)d_out);
}
